// HeterogeneousGAT_73246372266144
// MI455X (gfx1250) — matrix operand order repaired and checked by reading, not compiled, not run on hardware
//
#include <hip/hip_runtime.h>

// ---------------------------------------------------------------------------
// HeterogeneousGAT on gfx1250 (MI455X).
//   - GEMMs: v_wmma_f32_16x16x32_f16. Weights pre-converted once per call to
//     f16 in LDS-slab layout; double-buffered LDS staging (1 barrier/k-step)
//     using GLOBAL_LOAD_ASYNC_TO_LDS_B128 (ASYNCcnt-tracked),
//     f16 activations (A read once per row, two b128 loads per k-step).
//   - Graph part: deterministic CSR-by-dst gather softmax/aggregation,
//     no float atomics, float4 traffic.
// ---------------------------------------------------------------------------

typedef __attribute__((ext_vector_type(16))) _Float16 v16h;
typedef __attribute__((ext_vector_type(8)))  _Float16 v8h;
typedef __attribute__((ext_vector_type(8)))  float    v8f;
typedef __attribute__((ext_vector_type(4)))  float    f32x4;
typedef int v4i __attribute__((vector_size(16)));   // matches builtin's V4i

#if defined(__has_builtin)
#if __has_builtin(__builtin_amdgcn_global_load_async_to_lds_b128)
#define HAVE_ASYNC_COPY 1
#endif
#if __has_builtin(__builtin_amdgcn_s_wait_asynccnt)
#define HAVE_WAIT_ASYNC 1
#endif
#endif

constexpr int THREADS = 256;            // 8 waves (wave32) per block
constexpr int HEADS   = 4;
constexpr int HID     = 64;
constexpr int HC      = HEADS * HID;    // 256
constexpr float NEG_SLOPE = 0.2f;
constexpr int SCAN_B  = 256;            // elements per scan block
constexpr int BPAD    = 40;             // halves per slab column (80B, 16B-aligned)

// ------------------------------ small utils --------------------------------

__global__ void k_fill_i32(int* __restrict__ p, int v, int n) {
  int i = blockIdx.x * blockDim.x + threadIdx.x;
  if (i < n) p[i] = v;
}

__global__ void k_copy_i32(const int* __restrict__ a, int* __restrict__ b, int n) {
  int i = blockIdx.x * blockDim.x + threadIdx.x;
  if (i < n) b[i] = a[i];
}

__global__ void k_f32_to_f16(const float* __restrict__ s, _Float16* __restrict__ d, int n) {
  int i = blockIdx.x * blockDim.x + threadIdx.x;
  if (i < n) d[i] = (_Float16)s[i];
}

// Weight [K,NC] f32 row-major -> f16 slab layout: slab s covers K [32s,32s+32),
// element (k,col) at Wt[((k>>5)*NC + col)*BPAD + (k&31)].
__global__ void k_prep_w(const float* __restrict__ W, _Float16* __restrict__ Wt,
                         int K, int NC, int ncs) {
  int idx = blockIdx.x * blockDim.x + threadIdx.x;
  if (idx >= K * NC) return;
  int k = idx >> ncs, col = idx & (NC - 1);
  Wt[((size_t)((k >> 5) * NC + col)) * BPAD + (k & 31)] = (_Float16)W[idx];
}

// ------------------------------ CSR build ----------------------------------
// Edge e in [0, Etot): e < E -> (src=ei[e], dst=ei[E+e]); else self-loop e-E.

__global__ void k_degree(const int* __restrict__ ei, int E, int Etot,
                         int* __restrict__ deg) {
  int e = blockIdx.x * blockDim.x + threadIdx.x;
  if (e >= Etot) return;
  int d = (e < E) ? ei[E + e] : (e - E);
  atomicAdd(&deg[d], 1);
}

__global__ void k_scan_block(const int* __restrict__ in, int* __restrict__ incl,
                             int* __restrict__ bsum, int n) {
  __shared__ int sh[SCAN_B];
  int i = blockIdx.x * SCAN_B + threadIdx.x;
  sh[threadIdx.x] = (i < n) ? in[i] : 0;
  __syncthreads();
  for (int o = 1; o < SCAN_B; o <<= 1) {
    int t = (threadIdx.x >= (unsigned)o) ? sh[threadIdx.x - o] : 0;
    __syncthreads();
    sh[threadIdx.x] += t;
    __syncthreads();
  }
  if (i < n) incl[i] = sh[threadIdx.x];
  if (threadIdx.x == SCAN_B - 1) bsum[blockIdx.x] = sh[threadIdx.x];
}

// single block; nb <= 1024 (N=50000 -> nb=196); exclusive scan in place
__global__ void k_scan_bsums(int* __restrict__ bsum, int nb) {
  __shared__ int sh[1024];
  int t = threadIdx.x;
  int orig = (t < nb) ? bsum[t] : 0;
  sh[t] = orig;
  __syncthreads();
  for (int o = 1; o < 1024; o <<= 1) {
    int v = (t >= o) ? sh[t - o] : 0;
    __syncthreads();
    sh[t] += v;
    __syncthreads();
  }
  if (t < nb) bsum[t] = sh[t] - orig;
}

__global__ void k_finalize_rowptr(const int* __restrict__ incl, const int* __restrict__ boff,
                                  int* __restrict__ row_ptr, int n) {
  int i = blockIdx.x * blockDim.x + threadIdx.x;
  if (i < n) row_ptr[i + 1] = incl[i] + boff[i >> 8];   // SCAN_B == 256
  if (i == 0) row_ptr[0] = 0;
}

__global__ void k_fill_csr(const int* __restrict__ ei, int E, int Etot,
                           int* __restrict__ cursor, int* __restrict__ eid) {
  int e = blockIdx.x * blockDim.x + threadIdx.x;
  if (e >= Etot) return;
  int d = (e < E) ? ei[E + e] : (e - E);
  int pos = atomicAdd(&cursor[d], 1);
  eid[pos] = e;
}

// deterministic edge order inside each segment (avg degree ~17)
__global__ void k_sort_segments(const int* __restrict__ row_ptr, int* __restrict__ eid, int n) {
  int node = blockIdx.x * blockDim.x + threadIdx.x;
  if (node >= n) return;
  int beg = row_ptr[node], end = row_ptr[node + 1];
  for (int i = beg + 1; i < end; ++i) {
    int v = eid[i];
    int j = i - 1;
    while (j >= beg && eid[j] > v) { eid[j + 1] = eid[j]; --j; }
    eid[j + 1] = v;
  }
}

// ------------------------- WMMA GEMM  C = A @ B ----------------------------
// A: [M,K] f16 row-major. Bt: f16 slab layout (k_prep_w). C: [M, NT*16] f32.
// One wave -> full 16-row x (NT*16)-col tile; B double-buffered in LDS via
// async global->LDS copies (fallback: plain load + ds_store).
template <int NT>
__global__ void k_gemm_wmma(const _Float16* __restrict__ A, const _Float16* __restrict__ Bt,
                            float* __restrict__ C, int M, int K) {
  constexpr int NC = NT * 16;
  constexpr int CHUNKS = NC * BPAD / 8;          // 16-byte chunks per slab
  __shared__ _Float16 Bs[2][NC * BPAD];          // NT=16: 40 KB

  const int lane = threadIdx.x & 31;
  const int rtiles = M >> 4;
  int wt = blockIdx.x * (blockDim.x >> 5) + (threadIdx.x >> 5);
  if (wt >= rtiles) wt = rtiles - 1;             // duplicate work; uniform barriers
  const int half = lane >> 4;                    // 0: lanes 0-15, 1: lanes 16-31
  const int l16  = lane & 15;
  const int rbase = wt << 4;
  const int kofs = half << 3;                    // A half-lane K offset
  const _Float16* arow = A + (size_t)(rbase + l16) * K;

  v8f acc[NT] = {};
  const int nslab = K >> 5;

  auto stage = [&](int s, int buf) {
#if defined(HAVE_ASYNC_COPY)
    __attribute__((address_space(1))) v4i* src =
        (__attribute__((address_space(1))) v4i*)(Bt + (size_t)s * NC * BPAD);
    __attribute__((address_space(3))) v4i* dst =
        (__attribute__((address_space(3))) v4i*)(&Bs[buf][0]);
    for (int i = threadIdx.x; i < CHUNKS; i += THREADS)
      __builtin_amdgcn_global_load_async_to_lds_b128(src + i, dst + i, 0, 0);
#else
    const v8h* src = (const v8h*)(Bt + (size_t)s * NC * BPAD);
    v8h* dst = (v8h*)Bs[buf];
    for (int i = threadIdx.x; i < CHUNKS; i += THREADS) dst[i] = src[i];
#endif
  };

  auto stage_wait = [&]() {
#if defined(HAVE_ASYNC_COPY)
#if defined(HAVE_WAIT_ASYNC)
    __builtin_amdgcn_s_wait_asynccnt(0);
#else
    asm volatile("s_wait_asynccnt 0x0" ::: "memory");
#endif
#endif
  };

  stage(0, 0);
  stage_wait();
  __syncthreads();

  for (int s = 0; s < nslab; ++s) {
    const int cur = s & 1;
    if (s + 1 < nslab) stage(s + 1, cur ^ 1);    // overlaps compute of slab s

    // A fragment 16x32: half 0 -> K {0..7,16..23}, half 1 -> {8..15,24..31}
    const int kb = s << 5;
    v8h a0 = *(const v8h*)(arow + kb + kofs);
    v8h a1 = *(const v8h*)(arow + kb + 16 + kofs);
    v16h af = __builtin_shufflevector(a0, a1, 0, 1, 2, 3, 4, 5, 6, 7,
                                      8, 9, 10, 11, 12, 13, 14, 15);

#pragma unroll
    for (int nt = 0; nt < NT; ++nt) {
      // B fragment 32x16: half 0 -> K {0..7,16..23}, half 1 -> {8..15,24..31}
      const _Float16* bp = Bs[cur] + (nt * 16 + l16) * BPAD + (half << 3);
      v8h b0 = *(const v8h*)bp;
      v8h b1 = *(const v8h*)(bp + 16);
      v16h bf = __builtin_shufflevector(b0, b1, 0, 1, 2, 3, 4, 5, 6, 7,
                                        8, 9, 10, 11, 12, 13, 14, 15);
      acc[nt] = __builtin_amdgcn_wmma_f32_16x16x32_f16(
          false, af, false, bf, (short)0, acc[nt], false, false);
    }
    stage_wait();                                // our async fills done
    __syncthreads();                             // everyone's fills + reads done
  }

  // D layout: VGPR j -> M = j + 8*half, N = l16
#pragma unroll
  for (int nt = 0; nt < NT; ++nt)
#pragma unroll
    for (int j = 0; j < 8; ++j)
      C[(size_t)(rbase + j + (half << 3)) * NC + nt * 16 + l16] = acc[nt][j];
}

// -------- projection epilogue: + proj_b + type_emb, emit f16 for gemm0 ------
__global__ void k_proj_epilogue(const float* __restrict__ t, const float* __restrict__ proj_b,
                                const float* __restrict__ type_emb,
                                const int* __restrict__ node_types,
                                _Float16* __restrict__ xh, int n) {
  int i = blockIdx.x * blockDim.x + threadIdx.x;
  if (i >= n * HID) return;
  int node = i >> 6, c = i & 63;
  xh[i] = (_Float16)(t[i] + proj_b[c] + type_emb[node_types[node] * HID + c]);
}

// -------------- per-(node,head) attention dot products ----------------------
__global__ void k_attn_scores(const float* __restrict__ h,
                              const float* __restrict__ att_src,
                              const float* __restrict__ att_dst,
                              float* __restrict__ as_, float* __restrict__ ad_, int n) {
  int idx = blockIdx.x * blockDim.x + threadIdx.x;
  if (idx >= n * HEADS) return;
  int node = idx >> 2, hh = idx & 3;
  const float* row = h + (size_t)node * HC + hh * HID;
  const float* av  = att_src + hh * HID;
  const float* dv  = att_dst + hh * HID;
  f32x4 s4 = {}, d4 = {};
#pragma unroll 4
  for (int c = 0; c < HID; c += 4) {
    f32x4 v = *(const f32x4*)(row + c);
    s4 += v * *(const f32x4*)(av + c);
    d4 += v * *(const f32x4*)(dv + c);
  }
  as_[idx] = s4[0] + s4[1] + s4[2] + s4[3];
  ad_[idx] = d4[0] + d4[1] + d4[2] + d4[3];
}

// ------------- segment softmax over incoming edges (CSR gather) -------------
__global__ void k_softmax_csr(const int* __restrict__ row_ptr, const int* __restrict__ eid,
                              const int* __restrict__ ei, int E,
                              const float* __restrict__ as_, const float* __restrict__ ad_,
                              float* __restrict__ ea, int n) {
  int idx = blockIdx.x * blockDim.x + threadIdx.x;
  if (idx >= n * HEADS) return;
  int node = idx >> 2, hh = idx & 3;
  int beg = row_ptr[node], end = row_ptr[node + 1];
  float adn = ad_[node * HEADS + hh];
  float mx = -3.0e38f;
  for (int p = beg; p < end; ++p) {
    int e = eid[p];
    int s = (e < E) ? ei[e] : (e - E);
    float a = as_[s * HEADS + hh] + adn;
    a = (a > 0.f) ? a : NEG_SLOPE * a;           // leaky_relu
    ea[p * HEADS + hh] = a;
    mx = fmaxf(mx, a);
  }
  float sum = 0.f;
  for (int p = beg; p < end; ++p) {
    float ev = __expf(ea[p * HEADS + hh] - mx);
    ea[p * HEADS + hh] = ev;
    sum += ev;
  }
  float inv = 1.f / sum;                         // self-loop => sum > 0
  for (int p = beg; p < end; ++p) ea[p * HEADS + hh] *= inv;
}

// ---------- weighted aggregation (gather, no atomics, float4 lanes) ---------
__global__ void k_aggregate(const int* __restrict__ row_ptr, const int* __restrict__ eid,
                            const int* __restrict__ ei, int E,
                            const float* __restrict__ h, const float* __restrict__ alpha,
                            float* __restrict__ out, int n, int mean_mode) {
  int idx = blockIdx.x * blockDim.x + threadIdx.x;
  if (idx >= n * 16) return;                     // 16 x float4 channel groups
  int node = idx >> 4, c = (idx & 15) << 2;
  int beg = row_ptr[node], end = row_ptr[node + 1];
  f32x4 a0 = {}, a1 = {}, a2 = {}, a3 = {};
  for (int p = beg; p < end; ++p) {
    int e = eid[p];
    int s = (e < E) ? ei[e] : (e - E);
    const float* hs = h + (size_t)s * HC + c;
    f32x4 al = *(const f32x4*)(alpha + p * HEADS);
    a0 += *(const f32x4*)(hs)       * al[0];
    a1 += *(const f32x4*)(hs + 64)  * al[1];
    a2 += *(const f32x4*)(hs + 128) * al[2];
    a3 += *(const f32x4*)(hs + 192) * al[3];
  }
  if (mean_mode) {
    *(f32x4*)(out + (size_t)node * HID + c) = (a0 + a1 + a2 + a3) * 0.25f;
  } else {
    float* od = out + (size_t)node * HC + c;
    *(f32x4*)(od)       = a0;
    *(f32x4*)(od + 64)  = a1;
    *(f32x4*)(od + 128) = a2;
    *(f32x4*)(od + 192) = a3;
  }
}

// -------- (x + bias) -> LayerNorm -> ReLU; optional f32 and f16 outputs -----
template <int C>
__global__ void k_ln_relu(const float* __restrict__ in, const float* __restrict__ bias,
                          const float* __restrict__ g, const float* __restrict__ b,
                          float* __restrict__ out_f, _Float16* __restrict__ out_h, int n) {
  constexpr int PER = C >> 5;
  int wave = blockIdx.x * (blockDim.x >> 5) + (threadIdx.x >> 5);
  if (wave >= n) return;
  int lane = threadIdx.x & 31;
  const float* row = in + (size_t)wave * C;
  float vals[PER];
  float sum = 0.f;
#pragma unroll
  for (int i = 0; i < PER; ++i) {
    int c = lane + (i << 5);
    float v = row[c] + bias[c];
    vals[i] = v; sum += v;
  }
#pragma unroll
  for (int o = 16; o > 0; o >>= 1) sum += __shfl_xor(sum, o, 32);
  float mu = sum / (float)C;
  float var = 0.f;
#pragma unroll
  for (int i = 0; i < PER; ++i) { float t = vals[i] - mu; var += t * t; }
#pragma unroll
  for (int o = 16; o > 0; o >>= 1) var += __shfl_xor(var, o, 32);
  var /= (float)C;
  float rstd = rsqrtf(var + 1e-5f);
#pragma unroll
  for (int i = 0; i < PER; ++i) {
    int c = lane + (i << 5);
    float y = fmaxf((vals[i] - mu) * rstd * g[c] + b[c], 0.f);
    if (out_f) out_f[(size_t)wave * C + c] = y;
    if (out_h) out_h[(size_t)wave * C + c] = (_Float16)y;
  }
}

// ------------------------------- launcher ----------------------------------

static inline int cdiv(long long a, int b) { return (int)((a + b - 1) / b); }

extern "C" void kernel_launch(void* const* d_in, const int* in_sizes, int n_in,
                              void* d_out, int out_size, void* d_ws, size_t ws_size,
                              hipStream_t stream) {
  const float* node_features = (const float*)d_in[0];
  const int*   node_types    = (const int*)  d_in[1];
  const int*   edge_index    = (const int*)  d_in[2];
  const float* type_emb      = (const float*)d_in[3];
  const float* proj_w        = (const float*)d_in[4];
  const float* proj_b        = (const float*)d_in[5];
  const float* lin_w0        = (const float*)d_in[6];
  const float* att_src0      = (const float*)d_in[7];
  const float* att_dst0      = (const float*)d_in[8];
  const float* bias0         = (const float*)d_in[9];
  const float* ln_g0         = (const float*)d_in[10];
  const float* ln_b0         = (const float*)d_in[11];
  const float* lin_w1        = (const float*)d_in[12];
  const float* att_src1      = (const float*)d_in[13];
  const float* att_dst1      = (const float*)d_in[14];
  const float* bias1         = (const float*)d_in[15];
  const float* ln_g1         = (const float*)d_in[16];
  const float* ln_b1         = (const float*)d_in[17];

  const int Nn   = in_sizes[1];
  const int E    = in_sizes[2] / 2;
  const int Etot = E + Nn;
  const int FIN  = in_sizes[0] / Nn;             // 128
  const int NB   = cdiv(Nn, SCAN_B);             // scan blocks (<=1024)

  // ---- workspace carving: f32 region, then f16 region, then int region ----
  float* ws   = (float*)d_ws;
  float* agg1 = ws;                               // [N,64]  layer-1 aggregate
  float* h0   = agg1 + (size_t)Nn * HID;          // [N,256] h (both layers)
  float* acc0 = h0   + (size_t)Nn * HC;           // [N,256] proj tmp / layer-0 agg
  float* as_  = acc0 + (size_t)Nn * HC;           // [N,4]
  float* ad_  = as_  + (size_t)Nn * HEADS;        // [N,4]
  float* ea   = ad_  + (size_t)Nn * HEADS;        // [Etot,4]
  _Float16* ah      = (_Float16*)(ea + (size_t)Etot * HEADS);  // [N,256] acts f16
  _Float16* wt_proj = ah      + (size_t)Nn * HC;  // (128/32)*64*BPAD
  _Float16* wt0     = wt_proj + (size_t)(128 / 32) * 64 * BPAD;
  _Float16* wt1     = wt0     + (size_t)(64 / 32) * 256 * BPAD;
  int*      ibase   = (int*)(wt1 + (size_t)(256 / 32) * 256 * BPAD);
  int* deg     = ibase;                           // [N]
  int* cursor  = deg     + Nn;                    // [N]
  int* row_ptr = cursor  + Nn;                    // [N+1]
  int* incl    = row_ptr + (Nn + 1);              // [N]
  int* bsum    = incl    + Nn;                    // [NB]
  int* eid     = bsum    + NB;                    // [Etot]

  const int gw = cdiv(Nn >> 4, THREADS / 32);     // gemm blocks (row tiles / 8)

  // ------------------------- CSR build (by dst) -------------------------
  k_fill_i32<<<cdiv(Nn, THREADS), THREADS, 0, stream>>>(deg, 0, Nn);
  k_degree<<<cdiv(Etot, THREADS), THREADS, 0, stream>>>(edge_index, E, Etot, deg);
  k_scan_block<<<NB, SCAN_B, 0, stream>>>(deg, incl, bsum, Nn);
  k_scan_bsums<<<1, 1024, 0, stream>>>(bsum, NB);
  k_finalize_rowptr<<<cdiv(Nn, THREADS), THREADS, 0, stream>>>(incl, bsum, row_ptr, Nn);
  k_copy_i32<<<cdiv(Nn, THREADS), THREADS, 0, stream>>>(row_ptr, cursor, Nn);
  k_fill_csr<<<cdiv(Etot, THREADS), THREADS, 0, stream>>>(edge_index, E, Etot, cursor, eid);
  k_sort_segments<<<cdiv(Nn, THREADS), THREADS, 0, stream>>>(row_ptr, eid, Nn);

  // ---------------- one-time per call data preparation ----------------
  k_f32_to_f16<<<cdiv((long long)Nn * FIN, THREADS), THREADS, 0, stream>>>(
      node_features, ah, Nn * FIN);              // ah = node_features f16 [N,128]
  k_prep_w<<<cdiv(FIN * HID, THREADS), THREADS, 0, stream>>>(proj_w, wt_proj, FIN, HID, 6);
  k_prep_w<<<cdiv(HID * HC, THREADS), THREADS, 0, stream>>>(lin_w0, wt0, HID, HC, 8);
  k_prep_w<<<cdiv(HC * HC, THREADS), THREADS, 0, stream>>>(lin_w1, wt1, HC, HC, 8);

  // ---- x0 = X @ proj_w + proj_b + type_emb[types]  (f16 into ah) ----
  k_gemm_wmma<4><<<gw, THREADS, 0, stream>>>(ah, wt_proj, acc0, Nn, FIN);
  k_proj_epilogue<<<cdiv((long long)Nn * HID, THREADS), THREADS, 0, stream>>>(
      acc0, proj_b, type_emb, node_types, ah, Nn);

  const int nhg = cdiv((long long)Nn * HEADS, THREADS);
  const int ngg = cdiv((long long)Nn * 16, THREADS);
  const int nlw = cdiv(Nn, THREADS / 32);

  // ---------------------------- GAT layer 0 ----------------------------
  k_gemm_wmma<16><<<gw, THREADS, 0, stream>>>(ah, wt0, h0, Nn, HID);
  k_attn_scores<<<nhg, THREADS, 0, stream>>>(h0, att_src0, att_dst0, as_, ad_, Nn);
  k_softmax_csr<<<nhg, THREADS, 0, stream>>>(row_ptr, eid, edge_index, E, as_, ad_, ea, Nn);
  k_aggregate<<<ngg, THREADS, 0, stream>>>(row_ptr, eid, edge_index, E, h0, ea, acc0, Nn, 0);
  k_ln_relu<HC><<<nlw, THREADS, 0, stream>>>(acc0, bias0, ln_g0, ln_b0,
                                             (float*)nullptr, ah, Nn);  // x1 f16

  // ---------------------------- GAT layer 1 ----------------------------
  k_gemm_wmma<16><<<gw, THREADS, 0, stream>>>(ah, wt1, h0, Nn, HC);
  k_attn_scores<<<nhg, THREADS, 0, stream>>>(h0, att_src1, att_dst1, as_, ad_, Nn);
  k_softmax_csr<<<nhg, THREADS, 0, stream>>>(row_ptr, eid, edge_index, E, as_, ad_, ea, Nn);
  k_aggregate<<<ngg, THREADS, 0, stream>>>(row_ptr, eid, edge_index, E, h0, ea, agg1, Nn, 1);
  k_ln_relu<HID><<<nlw, THREADS, 0, stream>>>(agg1, bias1, ln_g1, ln_b1,
                                              (float*)d_out, (_Float16*)nullptr, Nn);
}
